// MultiHeadAttention_79637283602476
// MI455X (gfx1250) — matrix operand order repaired and checked by reading, not compiled, not run on hardware
//
#include <hip/hip_runtime.h>

// ---------------------------------------------------------------------------
// MHA for MI455X (gfx1250): all matmuls via v_wmma_f32_16x16x32_bf16 (wave32).
// Pipeline: f32->bf16 convert -> 3x WMMA GEMM (TDM-fed, double-buffered LDS)
//           -> flash-attention (WMMA QK^T + online softmax + WMMA PV)
//           -> WMMA GEMM O-projection (fp32 out).
// ---------------------------------------------------------------------------

typedef __bf16 bf16;
typedef __attribute__((ext_vector_type(16))) __bf16 v16bf;
typedef __attribute__((ext_vector_type(8)))  __bf16 v8bf;
typedef __attribute__((ext_vector_type(4)))  __bf16 v4bf;
typedef __attribute__((ext_vector_type(8)))  float  v8f;
typedef __attribute__((ext_vector_type(4)))  unsigned int u32x4;
typedef __attribute__((ext_vector_type(8)))  unsigned int u32x8;

#define NTOK 4096   // B*S
#define DM   1024
#define SEQ  2048
#define NH   16
#define HD   64

static __device__ __forceinline__ bf16 f2bf(float f) {
  unsigned int u = __builtin_bit_cast(unsigned int, f);
  u += 0x7fffu + ((u >> 16) & 1u);              // round-to-nearest-even
  unsigned short h = (unsigned short)(u >> 16);
  return __builtin_bit_cast(bf16, h);
}

static __device__ __forceinline__ v8f vzero8() {
  v8f z;
#pragma unroll
  for (int i = 0; i < 8; ++i) z[i] = 0.0f;
  return z;
}

// A-fragment (16x32 bf16, row-major source). Lane holds row=lane&15.
// K pattern per ISA: lanes 0-15 -> {0..7,16..23}; lanes 16-31 -> {8..15,24..31}.
static __device__ __forceinline__ v16bf load_fragA(const bf16* rowptr, int sel) {
  v8bf lo = *(const v8bf*)(rowptr + sel * 8);
  v8bf hi = *(const v8bf*)(rowptr + sel * 8 + 16);
  return __builtin_shufflevector(lo, hi, 0,1,2,3,4,5,6,7,8,9,10,11,12,13,14,15);
}

// B-fragment (32x16 bf16) built from row-major B^T (row = output col n=lane&15).
// K pattern per ISA: lanes 0-15 -> {0..7,16..23}; lanes 16-31 -> {8..15,24..31}.
static __device__ __forceinline__ v16bf load_fragB(const bf16* rowptr, int sel) {
  v8bf lo = *(const v8bf*)(rowptr + sel * 8);
  v8bf hi = *(const v8bf*)(rowptr + sel * 8 + 16);
  return __builtin_shufflevector(lo, hi, 0,1,2,3,4,5,6,7,8,9,10,11,12,13,14,15);
}

static __device__ __forceinline__ v8f wmma_bf16(v16bf a, v16bf b, v8f c) {
  return __builtin_amdgcn_wmma_f32_16x16x32_bf16(
      false, a, false, b, (short)0, c, false, false);
}

// ---------------------------------------------------------------------------
// Tensor Data Mover: 2D tile (global -> LDS), bf16 elements, rows packed
// contiguously into LDS. Descriptor per CDNA5 ISA:
//   group0: [1:0]=count=1, [63:32]=lds_addr, [120:64]=global_addr, [127:126]=2
//   group1: data_size=1(2B), tensor_dim0/1, tile_dim0/1, tensor_dim0_stride
// Issued wave-scalar via inline asm (toolchain-portable form).
// ---------------------------------------------------------------------------
static __device__ __forceinline__ void tdm_load_2d(unsigned lds_off, const void* gptr,
                                                   unsigned dim0, unsigned dim1,
                                                   unsigned stride0_elems) {
  unsigned long long ga = (unsigned long long)gptr;
  u32x4 g0;
  g0[0] = 1u;                                             // count=1 (valid)
  g0[1] = lds_off;                                        // LDS byte address
  g0[2] = (unsigned)(ga & 0xffffffffu);                   // global_addr[31:0]
  g0[3] = ((unsigned)(ga >> 32) & 0x01ffffffu) | (2u << 30); // [56:32] | type=2
  u32x8 g1;
  g1[0] = (1u << 16);                                     // wg_mask=0, data_size=1 (2B)
  g1[1] = (dim0 & 0xffffu) << 16;                         // tensor_dim0[15:0]
  g1[2] = (dim0 >> 16) | ((dim1 & 0xffffu) << 16);        // tensor_dim0 hi | dim1 lo
  g1[3] = (dim1 >> 16) | (dim0 << 16);                    // tensor_dim1 hi | tile_dim0
  g1[4] = dim1 & 0xffffu;                                 // tile_dim1 | tile_dim2=0
  g1[5] = stride0_elems;                                  // tensor_dim0_stride[31:0]
  g1[6] = 0u;                                             // stride0 hi | stride1 lo
  g1[7] = 0u;                                             // stride1 hi
  asm volatile("tensor_load_to_lds %0, %1" :: "s"(g0), "s"(g1) : "memory");
}

// ---------------------------------------------------------------------------
// f32 -> bf16 conversion (4 elems/thread)
// ---------------------------------------------------------------------------
__global__ __launch_bounds__(256)
void mha_cvt_bf16(const float* __restrict__ in, bf16* __restrict__ out, int n4) {
  int i = blockIdx.x * blockDim.x + threadIdx.x;
  if (i < n4) {
    float4 v = ((const float4*)in)[i];
    v4bf o;
    o[0] = f2bf(v.x); o[1] = f2bf(v.y); o[2] = f2bf(v.z); o[3] = f2bf(v.w);
    *(v4bf*)(out + 4 * i) = o;
  }
}

// ---------------------------------------------------------------------------
// GEMM: C[M=4096, N=1024] = A[M,1024] * W[1024,1024]^T + bias
// Block = 8 waves -> 128x64 tile. TDM double-buffers A(128x32) and W(64x32)
// tiles into LDS; waves consume fragments via DS loads (W shared by 8 waves).
// mode 0: bf16 row-major out[M,1024]
// mode 1: bf16 transposed out[1024, M]   (for V)
// mode 2: f32 row-major out[M,1024]      (final O projection)
// ---------------------------------------------------------------------------
__global__ __launch_bounds__(256)
void mha_gemm_bf16(const bf16* __restrict__ A, const bf16* __restrict__ W,
                   const float* __restrict__ bias, void* __restrict__ outp,
                   int mode) {
  __shared__ bf16 shA[2][128 * 32];   // 2 x 8KB
  __shared__ bf16 shW[2][64 * 32];    // 2 x 4KB

  const int lane = threadIdx.x & 31;
  const int row  = lane & 15;
  const int sel  = lane >> 4;
  const int wid  = threadIdx.x >> 5;            // 0..7
  const int mblk = (int)(blockIdx.x >> 4) * 128; // 32 M-blocks
  const int nblk = (int)(blockIdx.x & 15) * 64;  // 16 N-blocks

  if (wid == 0) {   // prime the TDM pipeline (buffer 0, k=0)
    tdm_load_2d((unsigned)(size_t)&shA[0][0], A + (size_t)mblk * DM, 32, 128, DM);
    tdm_load_2d((unsigned)(size_t)&shW[0][0], W + (size_t)nblk * DM, 32, 64, DM);
  }

  v8f acc[4];
#pragma unroll
  for (int t = 0; t < 4; ++t) acc[t] = vzero8();

  for (int i = 0; i < 32; ++i) {                // 32 K-steps of 32
    const int p = i & 1;
    if (wid == 0) {
      if (i + 1 < 32) {                         // prefetch next buffer via TDM
        const int k = (i + 1) * 32;
        tdm_load_2d((unsigned)(size_t)&shA[p ^ 1][0],
                    A + (size_t)mblk * DM + k, 32, 128, DM);
        tdm_load_2d((unsigned)(size_t)&shW[p ^ 1][0],
                    W + (size_t)nblk * DM + k, 32, 64, DM);
        __builtin_amdgcn_s_wait_tensorcnt(2);   // current buffer's 2 TDMs done
      } else {
        __builtin_amdgcn_s_wait_tensorcnt(0);
      }
    }
    __syncthreads();
    v16bf a = load_fragA(&shA[p][(wid * 16 + row) * 32], sel);
#pragma unroll
    for (int t = 0; t < 4; ++t) {
      v16bf b = load_fragB(&shW[p][(t * 16 + row) * 32], sel);
      acc[t] = wmma_bf16(a, b, acc[t]);
    }
    __syncthreads();
  }

  const int m0 = mblk + wid * 16;
#pragma unroll
  for (int t = 0; t < 4; ++t) {
    const int ng = nblk + t * 16 + row;         // output column (lane's N)
    const float bb = bias[ng];
    if (mode == 0) {
      bf16* out = (bf16*)outp;
#pragma unroll
      for (int r = 0; r < 8; ++r)
        out[(size_t)(m0 + sel * 8 + r) * DM + ng] = f2bf(acc[t][r] + bb);
    } else if (mode == 1) {
      bf16* out = (bf16*)outp;                  // [DM][NTOK] transposed
      v8bf o;
#pragma unroll
      for (int r = 0; r < 8; ++r) o[r] = f2bf(acc[t][r] + bb);
      *(v8bf*)(out + (size_t)ng * NTOK + m0 + sel * 8) = o;
    } else {
      float* out = (float*)outp;
#pragma unroll
      for (int r = 0; r < 8; ++r)
        out[(size_t)(m0 + sel * 8 + r) * DM + ng] = acc[t][r] + bb;
    }
  }
}

// ---------------------------------------------------------------------------
// Flash attention: one wave per (b, h, 16-query tile); 32 keys per step.
// Qb,Kb: bf16 [NTOK, DM] row-major.  Vt: bf16 [DM, NTOK] (transposed).
// ---------------------------------------------------------------------------
__global__ __launch_bounds__(128)
void mha_attn(const bf16* __restrict__ Qb, const bf16* __restrict__ Kb,
              const bf16* __restrict__ Vt, const int* __restrict__ mask,
              bf16* __restrict__ Ob) {
  __shared__ bf16 plds[4][16 * 32];         // per-wave P staging (D-layout -> A-layout)

  const int lane = threadIdx.x & 31;
  const int col  = lane & 15;
  const int sel  = lane >> 4;
  const int wl   = threadIdx.x >> 5;
  const int wave = blockIdx.x * 4 + wl;     // 0..4095 = B*NH*(SEQ/16)

  const int qi = wave & 127;
  const int h  = (wave >> 7) & 15;
  const int b  = wave >> 11;
  const int tok0 = b * SEQ + qi * 16;
  const int hc   = h * HD;

  const bf16* qrow = Qb + (size_t)(tok0 + col) * DM + hc;
  const v16bf q0 = load_fragA(qrow, sel);         // head-dim 0..31
  const v16bf q1 = load_fragA(qrow + 32, sel);    // head-dim 32..63

  float mrow[8], lrow[8];
  v8f acc[4];
#pragma unroll
  for (int r = 0; r < 8; ++r) { mrow[r] = -1e30f; lrow[r] = 0.0f; }
#pragma unroll
  for (int t = 0; t < 4; ++t) acc[t] = vzero8();

  bf16* pl = plds[wl];
  const int* mk = mask + b * SEQ;

  for (int jb = 0; jb < SEQ; jb += 32) {
    // ---- scores: two 16x16 tiles (keys jb..jb+15, jb+16..jb+31) ----
    const bf16* kr0 = Kb + (size_t)(b * SEQ + jb + col) * DM + hc;
    __builtin_prefetch(kr0 + (size_t)32 * DM, 0, 3);   // next K tile (rows 0-15)
    __builtin_prefetch(kr0 + (size_t)48 * DM, 0, 3);   // next K tile (rows 16-31)
    v8f s0 = vzero8(), s1 = vzero8();
    {
      s0 = wmma_bf16(q0, load_fragB(kr0, sel), s0);
      s0 = wmma_bf16(q1, load_fragB(kr0 + 32, sel), s0);
      const bf16* kr1 = kr0 + (size_t)16 * DM;
      s1 = wmma_bf16(q0, load_fragB(kr1, sel), s1);
      s1 = wmma_bf16(q1, load_fragB(kr1 + 32, sel), s1);
    }
    const int m0v = mk[jb + col];
    const int m1v = mk[jb + 16 + col];

    // ---- online softmax (rows striped across 16-lane halves) ----
#pragma unroll
    for (int r = 0; r < 8; ++r) {
      float a0 = m0v ? s0[r] * 0.125f : -1e30f;
      float a1 = m1v ? s1[r] * 0.125f : -1e30f;
      float v = fmaxf(a0, a1);
#pragma unroll
      for (int o = 1; o < 16; o <<= 1) v = fmaxf(v, __shfl_xor(v, o, 16));
      float mnew = fmaxf(mrow[r], v);
      float corr = __expf(mrow[r] - mnew);
      mrow[r] = mnew;
      float p0 = __expf(a0 - mnew);
      float p1 = __expf(a1 - mnew);
      float ps = p0 + p1;
#pragma unroll
      for (int o = 1; o < 16; o <<= 1) ps += __shfl_xor(ps, o, 16);
      lrow[r] = lrow[r] * corr + ps;
#pragma unroll
      for (int t = 0; t < 4; ++t) acc[t][r] *= corr;
      // stage P into LDS at [row m][local key kk]
      pl[(r + sel * 8) * 32 + col]      = f2bf(p0);
      pl[(r + sel * 8) * 32 + 16 + col] = f2bf(p1);
    }
    asm volatile("s_wait_dscnt 0" ::: "memory");  // same-wave DS in-order + barrier
    const v16bf pA = load_fragA(pl + col * 32, sel);

    // ---- PV: P(16x32) x V(32x64), V columns contiguous in Vt rows ----
    const bf16* vb = Vt + (size_t)(hc + col) * NTOK + b * SEQ + jb;
    __builtin_prefetch(vb + 32, 0, 3);            // next V keys
#pragma unroll
    for (int t = 0; t < 4; ++t)
      acc[t] = wmma_bf16(pA, load_fragB(vb + (size_t)t * 16 * NTOK, sel), acc[t]);
  }

  // ---- normalize + store ----
#pragma unroll
  for (int t = 0; t < 4; ++t) {
    const int ng = hc + t * 16 + col;
#pragma unroll
    for (int r = 0; r < 8; ++r) {
      float o = acc[t][r] / lrow[r];
      Ob[(size_t)(tok0 + sel * 8 + r) * DM + ng] = f2bf(o);
    }
  }
}

// ---------------------------------------------------------------------------
extern "C" void kernel_launch(void* const* d_in, const int* in_sizes, int n_in,
                              void* d_out, int out_size, void* d_ws, size_t ws_size,
                              hipStream_t stream) {
  const float* query = (const float*)d_in[0];
  const float* key   = (const float*)d_in[1];
  const float* value = (const float*)d_in[2];
  const int*   mask  = (const int*)d_in[3];
  const float* Wq = (const float*)d_in[4];  const float* bq = (const float*)d_in[5];
  const float* Wk = (const float*)d_in[6];  const float* bk = (const float*)d_in[7];
  const float* Wv = (const float*)d_in[8];  const float* bv = (const float*)d_in[9];
  const float* Wo = (const float*)d_in[10]; const float* bo = (const float*)d_in[11];

  // workspace layout (bf16 elements), total 64 MB
  bf16* xq = (bf16*)d_ws;
  bf16* xk = xq + (size_t)NTOK * DM;
  bf16* xv = xk + (size_t)NTOK * DM;
  bf16* wq = xv + (size_t)NTOK * DM;
  bf16* wk = wq + (size_t)DM * DM;
  bf16* wv = wk + (size_t)DM * DM;
  bf16* wo = wv + (size_t)DM * DM;
  bf16* Qb = wo + (size_t)DM * DM;
  bf16* Kb = Qb + (size_t)NTOK * DM;
  bf16* Vt = Kb + (size_t)NTOK * DM;
  bf16* Ab = Vt + (size_t)NTOK * DM;

  // 1) convert activations (4M elems) and weights (1M elems) to bf16
  mha_cvt_bf16<<<4096, 256, 0, stream>>>(query, xq, NTOK * DM / 4);
  mha_cvt_bf16<<<4096, 256, 0, stream>>>(key,   xk, NTOK * DM / 4);
  mha_cvt_bf16<<<4096, 256, 0, stream>>>(value, xv, NTOK * DM / 4);
  mha_cvt_bf16<<<1024, 256, 0, stream>>>(Wq, wq, DM * DM / 4);
  mha_cvt_bf16<<<1024, 256, 0, stream>>>(Wk, wk, DM * DM / 4);
  mha_cvt_bf16<<<1024, 256, 0, stream>>>(Wv, wv, DM * DM / 4);
  mha_cvt_bf16<<<1024, 256, 0, stream>>>(Wo, wo, DM * DM / 4);

  // 2) projections (512 blocks x 8 waves; TDM-fed 128x64 tiles)
  mha_gemm_bf16<<<512, 256, 0, stream>>>(xq, wq, bq, Qb, 0);
  mha_gemm_bf16<<<512, 256, 0, stream>>>(xk, wk, bk, Kb, 0);
  mha_gemm_bf16<<<512, 256, 0, stream>>>(xv, wv, bv, Vt, 1);   // transposed

  // 3) attention (B*NH*S/16 = 4096 waves)
  mha_attn<<<1024, 128, 0, stream>>>(Qb, Kb, Vt, mask, Ab);

  // 4) output projection -> fp32 d_out
  mha_gemm_bf16<<<512, 256, 0, stream>>>(Ab, wo, bo, d_out, 2);
}
